// ModularSelectCascadeNet_21921513079161
// MI455X (gfx1250) — matrix operand order repaired and checked by reading, not compiled, not run on hardware
//
#include <hip/hip_runtime.h>

typedef __attribute__((ext_vector_type(16))) _Float16 v16h;
typedef __attribute__((ext_vector_type(8)))  _Float16 v8h;
typedef __attribute__((ext_vector_type(4)))  _Float16 v4h;
typedef __attribute__((ext_vector_type(8)))  float    v8f;

// ---------------------------------------------------------------- utilities
__device__ __forceinline__ float gumbelize(float u) {
    u = fminf(fmaxf(u, 1e-10f), 1.0f - 1e-7f);
    return -__logf(-__logf(u));
}

// ------------------------------------------------- weight transpose/convert
// W: [Z][Kin][N] f32  ->  Wt: [Z][N][Kpad] f16 (zero-padded K)
__global__ void prep_weight_kernel(const float* __restrict__ W,
                                   _Float16* __restrict__ Wt,
                                   int Kin, int N, int Kpad, int total) {
    int idx = blockIdx.x * 256 + threadIdx.x;
    if (idx >= total) return;
    int kp = idx % Kpad;
    int n  = (idx / Kpad) % N;
    int z  = idx / (Kpad * N);
    float v = (kp < Kin) ? W[((size_t)z * Kin + kp) * N + n] : 0.0f;
    Wt[idx] = (_Float16)v;
}

// ------------------------------------------------------- generic WMMA GEMM
// C[z] = act( (normalize?(A[z]) @ Wt[z]^T) + bias[z] )   (A f32, W f16, acc f32)
// Block: 256 thr = 8 waves; block tile = 128 rows x (NT*16) cols.
// Each wave owns one 16-row slice and NT N-tiles. NT is compile-time:
// the K-loop has no control flow at all.
template<int NT, bool DONORM>
__global__ __launch_bounds__(256) void gemm_wmma_kernel(
    const float* __restrict__ A, int lda, long a_zs,
    const _Float16* __restrict__ Wt, long w_zs, int Kpad,
    const float* __restrict__ bias, int b_zs,
    float* __restrict__ C, int ldc, int c_zoff,
    int K, int nt0base, int act,
    const float* __restrict__ aux, int ldaux,
    const float* __restrict__ nmean, const float* __restrict__ nrstd,
    const float* __restrict__ ngamma, const float* __restrict__ nbeta, int g_zs)
{
    __shared__ __align__(16) _Float16 As[128 * 40];   // 40-halves row pitch
    const int tid    = threadIdx.x;
    const int wave   = tid >> 5;
    const int lane   = tid & 31;
    const int laneHi = lane >> 4;
    const int lane16 = lane & 15;
    const int z      = blockIdx.z;
    const int Mbase  = blockIdx.y * 128;
    const int nt0    = nt0base + blockIdx.x * NT;

    const float*    Ab = A  + (size_t)z * a_zs;
    const _Float16* Wz = Wt + (size_t)z * w_zs;

    // per-tile W column base pointers (hoisted out of the K-loop)
    const _Float16* wpn[NT];
    #pragma unroll
    for (int nt = 0; nt < NT; ++nt)
        wpn[nt] = Wz + (size_t)((nt0 + nt) * 16 + lane16) * Kpad + 8 * laneHi;

    v8f acc[NT];
    #pragma unroll
    for (int nt = 0; nt < NT; ++nt) acc[nt] = (v8f){};

    const int asBase = (wave * 16 + lane16) * 40;

    for (int kc = 0; kc < Kpad; kc += 32) {
        __syncthreads();
        // stage A chunk (128 x 32) as f16: one float4 -> half4 per thread-iter
        #pragma unroll
        for (int it = 0; it < 4; ++it) {
            int idx = tid + it * 256;          // 1024 float4 groups
            int r   = idx >> 3;
            int col = (idx & 7) * 4;
            int k   = kc + col;
            float4 v = make_float4(0.0f, 0.0f, 0.0f, 0.0f);
            if (k < K) {                        // K % 4 == 0 always
                v = *(const float4*)&Ab[(size_t)(Mbase + r) * lda + k];
                if (DONORM) {
                    const float4 m  = *(const float4*)&nmean[k];
                    const float4 rs = *(const float4*)&nrstd[k];
                    const float4 g  = *(const float4*)&ngamma[(size_t)z * g_zs + k];
                    const float4 be = *(const float4*)&nbeta[(size_t)z * g_zs + k];
                    v.x = (v.x - m.x) * rs.x * g.x + be.x;
                    v.y = (v.y - m.y) * rs.y * g.y + be.y;
                    v.z = (v.z - m.z) * rs.z * g.z + be.z;
                    v.w = (v.w - m.w) * rs.w * g.w + be.w;
                }
            }
            v4h hv;
            hv[0] = (_Float16)v.x; hv[1] = (_Float16)v.y;
            hv[2] = (_Float16)v.z; hv[3] = (_Float16)v.w;
            *(v4h*)&As[r * 40 + col] = hv;
        }
        __syncthreads();
        // A fragment: 16x32 f16 ISA layout (lanes 0-15: K e / e+16; hi: +8)
        union { v16h v; v8h h[2]; } af;
        af.h[0] = *(const v8h*)&As[asBase + 8 * laneHi];
        af.h[1] = *(const v8h*)&As[asBase + 16 + 8 * laneHi];
        // load ALL B fragments first (one wait), then back-to-back WMMAs
        v8h b0[NT], b1[NT];
        #pragma unroll
        for (int nt = 0; nt < NT; ++nt) {
            b0[nt] = *(const v8h*)(wpn[nt] + kc);
            b1[nt] = *(const v8h*)(wpn[nt] + kc + 16);
        }
        #pragma unroll
        for (int nt = 0; nt < NT; ++nt) {
            union { v16h v; v8h h[2]; } bf;
            bf.h[0] = b0[nt];
            bf.h[1] = b1[nt];
            acc[nt] = __builtin_amdgcn_wmma_f32_16x16x32_f16(
                          false, af.v, false, bf.v, (short)0, acc[nt], false, false);
        }
    }

    #pragma unroll
    for (int nt = 0; nt < NT; ++nt) {
        int ncol = (nt0 + nt) * 16 + lane16;
        float bv = bias ? bias[(size_t)z * b_zs + ncol] : 0.0f;
        #pragma unroll
        for (int r = 0; r < 8; ++r) {
            int grow = Mbase + wave * 16 + r + 8 * laneHi;  // C elem r <-> M=r+8*laneHi
            float v = acc[nt][r] + bv;
            if      (act == 1) v = fmaxf(v, 0.0f);
            else if (act == 2) v = tanhf(v);
            else if (act == 3) { v *= aux[(size_t)grow * ldaux + ncol]; v = fmaxf(v, 0.0f); }
            C[(size_t)grow * ldc + (size_t)z * c_zoff + ncol] = v;
        }
    }
}

// ------------------------------------------------------------- elementwise
__global__ void relu_copy_kernel(const float* __restrict__ in,
                                 float* __restrict__ out, long n) {
    long i = (long)blockIdx.x * 256 + threadIdx.x;
    if (i < n) out[i] = fmaxf(in[i], 0.0f);
}

// gumbel-softmax over rows of 8 (logit: Bx64, U/G: Bx3x64, layer i)
__global__ void gumbel8_kernel(const float* __restrict__ logit,
                               const float* __restrict__ U,
                               float* __restrict__ G, int i, int B) {
    int idx = blockIdx.x * 256 + threadIdx.x;   // b*8 + row
    if (idx >= B * 8) return;
    int b = idx >> 3, mrow = idx & 7;
    const float* lp = logit + (size_t)b * 64 + mrow * 8;
    const float* up = U + ((size_t)b * 3 + i) * 64 + mrow * 8;
    float v[8], mx = -1e30f;
    #pragma unroll
    for (int j = 0; j < 8; ++j) { v[j] = lp[j] + gumbelize(up[j]); mx = fmaxf(mx, v[j]); }
    float s = 0.0f;
    #pragma unroll
    for (int j = 0; j < 8; ++j) { v[j] = __expf(v[j] - mx); s += v[j]; }
    float inv = 1.0f / s;
    float* gp = G + ((size_t)b * 3 + i) * 64 + mrow * 8;
    #pragma unroll
    for (int j = 0; j < 8; ++j) gp[j] = v[j] * inv;
}

// final_select = gumbel_softmax(si @ selF_W + selF_b, u_final); wave per row
__global__ __launch_bounds__(256) void final_select_kernel(
    const float* __restrict__ si, const float* __restrict__ W,   // 256x8
    const float* __restrict__ b8, const float* __restrict__ uF,
    float* __restrict__ fsel, int B)
{
    int wave = threadIdx.x >> 5, lane = threadIdx.x & 31;
    int b = blockIdx.x * 8 + wave;
    if (b >= B) return;
    float acc[8] = {};
    const float* row = si + (size_t)b * 256;
    for (int k = lane; k < 256; k += 32) {
        float x = row[k];
        const float* wr = W + k * 8;
        #pragma unroll
        for (int o = 0; o < 8; ++o) acc[o] += x * wr[o];
    }
    #pragma unroll
    for (int o = 0; o < 8; ++o)
        for (int off = 16; off; off >>= 1) acc[o] += __shfl_down(acc[o], off, 32);
    if (lane == 0) {
        const float* up = uF + (size_t)b * 8;
        float v[8], mx = -1e30f;
        #pragma unroll
        for (int o = 0; o < 8; ++o) { v[o] = acc[o] + b8[o] + gumbelize(up[o]); mx = fmaxf(mx, v[o]); }
        float s = 0.0f;
        #pragma unroll
        for (int o = 0; o < 8; ++o) { v[o] = __expf(v[o] - mx); s += v[o]; }
        float inv = 1.0f / s;
        float* fp = fsel + (size_t)b * 8;
        #pragma unroll
        for (int o = 0; o < 8; ++o) fp[o] = v[o] * inv;
    }
}

// ------------------------------------- deterministic two-stage batch stats
__global__ __launch_bounds__(256) void stats_partial_kernel(
    const float* __restrict__ X, int ldx, int ncols,
    float* __restrict__ Sp, float* __restrict__ Qp)
{
    int rb = blockIdx.x * 64;
    for (int c = threadIdx.x; c < ncols; c += 256) {
        float s = 0.0f, q = 0.0f;
        for (int r = 0; r < 64; ++r) {
            float v = X[(size_t)(rb + r) * ldx + c];
            s += v; q += v * v;
        }
        Sp[(size_t)blockIdx.x * ncols + c] = s;
        Qp[(size_t)blockIdx.x * ncols + c] = q;
    }
}

__global__ void stats_final_kernel(const float* __restrict__ Sp,
                                   const float* __restrict__ Qp,
                                   int nblk, int ncols,
                                   float* __restrict__ mean,
                                   float* __restrict__ rstd, float invB) {
    int c = blockIdx.x * 256 + threadIdx.x;
    if (c >= ncols) return;
    float s = 0.0f, q = 0.0f;
    for (int b = 0; b < nblk; ++b) { s += Sp[(size_t)b * ncols + c]; q += Qp[(size_t)b * ncols + c]; }
    float m = s * invB;
    mean[c] = m;
    rstd[c] = rsqrtf(q * invB - m * m + 1e-5f);
}

// --------------------------- per-sample cascade: bn2 -> 3x (8x8 mix, relu)
// -> fsel contraction -> 256x18 head. One wave per sample; 8 cols per lane.
__global__ __launch_bounds__(256) void cascade_kernel(
    const float* __restrict__ h,      // B x 8 x 256
    const float* __restrict__ mean2, const float* __restrict__ rstd2,   // 2048
    const float* __restrict__ g2, const float* __restrict__ b2,         // 8x256
    const float* __restrict__ gsel,   // B x 3 x 64
    const float* __restrict__ fsel,   // B x 8
    const float* __restrict__ lastW,  // 256 x 18
    const float* __restrict__ lastb,  // 18
    float* __restrict__ out, int B)
{
    int wave = threadIdx.x >> 5, lane = threadIdx.x & 31;
    int b = blockIdx.x * 8 + wave;
    if (b >= B) return;

    float prev[8][8];
    #pragma unroll
    for (int m = 0; m < 8; ++m)
        #pragma unroll
        for (int j = 0; j < 8; ++j) {
            int c = m * 256 + lane + 32 * j;
            float v = h[(size_t)b * 2048 + c];
            prev[m][j] = (v - mean2[c]) * rstd2[c] * g2[c] + b2[c];
        }

    // selects = gsels[::-1]  =>  apply layer 2, then 1, then 0
    for (int li = 2; li >= 0; --li) {
        const float* sp = gsel + ((size_t)b * 3 + li) * 64;
        float nxt[8][8];
        #pragma unroll
        for (int m = 0; m < 8; ++m) {
            float sk[8];
            #pragma unroll
            for (int k = 0; k < 8; ++k) sk[k] = sp[m * 8 + k];
            #pragma unroll
            for (int j = 0; j < 8; ++j) {
                float a = 0.0f;
                #pragma unroll
                for (int k = 0; k < 8; ++k) a += sk[k] * prev[k][j];
                nxt[m][j] = fmaxf(a, 0.0f);
            }
        }
        #pragma unroll
        for (int m = 0; m < 8; ++m)
            #pragma unroll
            for (int j = 0; j < 8; ++j) prev[m][j] = nxt[m][j];
    }

    const float* fp = fsel + (size_t)b * 8;
    float fs[8];
    #pragma unroll
    for (int m = 0; m < 8; ++m) fs[m] = fp[m];
    float outv[8];
    #pragma unroll
    for (int j = 0; j < 8; ++j) {
        float a = 0.0f;
        #pragma unroll
        for (int m = 0; m < 8; ++m) a += fs[m] * prev[m][j];
        outv[j] = a;
    }

    float y[18];
    #pragma unroll
    for (int o = 0; o < 18; ++o) y[o] = 0.0f;
    #pragma unroll
    for (int j = 0; j < 8; ++j) {
        const float* wr = lastW + (size_t)(lane + 32 * j) * 18;
        #pragma unroll
        for (int o = 0; o < 18; ++o) y[o] += outv[j] * wr[o];
    }
    #pragma unroll
    for (int o = 0; o < 18; ++o)
        for (int off = 16; off; off >>= 1) y[o] += __shfl_down(y[o], off, 32);
    if (lane == 0) {
        float* op = out + (size_t)b * 18;
        #pragma unroll
        for (int o = 0; o < 18; ++o) op[o] = y[o] + lastb[o];
    }
}

// --------------------------------------------------------------- launcher
extern "C" void kernel_launch(void* const* d_in, const int* in_sizes, int n_in,
                              void* d_out, int out_size, void* d_ws, size_t ws_size,
                              hipStream_t stream) {
    const float* x      = (const float*)d_in[0];
    const float* embi   = (const float*)d_in[1];
    const float* u_sel  = (const float*)d_in[2];
    const float* u_fin  = (const float*)d_in[3];
    const float* base_W0= (const float*)d_in[4];
    const float* base_b0= (const float*)d_in[5];
    const float* base_W1= (const float*)d_in[6];
    const float* base_b1= (const float*)d_in[7];
    const float* em_W0  = (const float*)d_in[8];
    const float* em_b0  = (const float*)d_in[9];
    const float* gat_W0 = (const float*)d_in[10];
    const float* gat_b0 = (const float*)d_in[11];
    const float* gat_W1 = (const float*)d_in[12];
    const float* gat_b1 = (const float*)d_in[13];
    const float* sel_W  = (const float*)d_in[14];
    const float* sel_b  = (const float*)d_in[15];
    const float* selF_W = (const float*)d_in[16];
    const float* selF_b = (const float*)d_in[17];
    const float* cond_W = (const float*)d_in[18];
    const float* cond_b = (const float*)d_in[19];
    const float* mod_W  = (const float*)d_in[20];
    const float* mod_b  = (const float*)d_in[21];
    const float* last_W = (const float*)d_in[22];
    const float* last_b = (const float*)d_in[23];
    const float* bn1_g  = (const float*)d_in[24];
    const float* bn1_b  = (const float*)d_in[25];
    const float* bn2_g  = (const float*)d_in[26];
    const float* bn2_b  = (const float*)d_in[27];

    const int B = in_sizes[0] / 128;   // 16384
    const int nblk = B / 64;

    char* ws = (char*)d_ws;
    size_t cur = 0;
    auto carve = [&](size_t bytes) -> char* {
        char* p = ws + cur;
        cur += (bytes + 255) & ~(size_t)255;
        return p;
    };
    _Float16* WtB0 = (_Float16*)carve((size_t)400*128*2);
    _Float16* WtB1 = (_Float16*)carve((size_t)256*416*2);
    _Float16* WtE0 = (_Float16*)carve((size_t)400*128*2);
    _Float16* WtG0 = (_Float16*)carve((size_t)256*416*2);
    _Float16* WtG1 = (_Float16*)carve((size_t)256*256*2);
    _Float16* WtS  = (_Float16*)carve((size_t)3*64*256*2);
    _Float16* WtC  = (_Float16*)carve((size_t)3*256*64*2);
    _Float16* WtM  = (_Float16*)carve((size_t)8*256*256*2);
    float* t0    = (float*)carve((size_t)B*400*4);
    float* outb  = (float*)carve((size_t)B*256*4);
    float* t2    = (float*)carve((size_t)B*256*4);
    float* embv  = (float*)carve((size_t)B*256*4);
    float* si    = (float*)carve((size_t)B*256*4);
    float* logit = (float*)carve((size_t)B*64*4);
    float* gsel  = (float*)carve((size_t)B*192*4);
    float* fsel  = (float*)carve((size_t)B*8*4);
    float* hbuf  = (float*)carve((size_t)B*2048*4);
    float* Sp    = (float*)carve((size_t)nblk*2048*4);
    float* Qp    = (float*)carve((size_t)nblk*2048*4);
    float* mean1 = (float*)carve(256*4);
    float* rstd1 = (float*)carve(256*4);
    float* mean2 = (float*)carve(2048*4);
    float* rstd2 = (float*)carve(2048*4);

    dim3 blk(256);
    auto prep = [&](const float* W, _Float16* Wt, int Kin, int N, int Kpad, int Z) {
        int total = Z * N * Kpad;
        prep_weight_kernel<<<dim3((total + 255) / 256), blk, 0, stream>>>(W, Wt, Kin, N, Kpad, total);
    };
    auto gemm = [&](const float* A, int lda, long azs,
                    const _Float16* Wt, long wzs, int K, int Kpad,
                    const float* bias, int bzs,
                    float* C, int ldc, int czoff, int N, int Z, int act,
                    const float* aux, int ldaux,
                    const float* nm, const float* nr,
                    const float* ng, const float* nb, int gzs) {
        const int Ntiles = N / 16;
        const int full   = Ntiles / 4;
        const int rem    = Ntiles % 4;
        const int Mb     = B / 128;
        #define LAUNCH_GEMM(NTV, GX, NT0)                                            \
            do {                                                                     \
                dim3 grid_((GX), Mb, Z);                                             \
                if (nm) gemm_wmma_kernel<NTV, true><<<grid_, blk, 0, stream>>>(      \
                    A, lda, azs, Wt, wzs, Kpad, bias, bzs, C, ldc, czoff, K, (NT0),  \
                    act, aux, ldaux, nm, nr, ng, nb, gzs);                           \
                else gemm_wmma_kernel<NTV, false><<<grid_, blk, 0, stream>>>(        \
                    A, lda, azs, Wt, wzs, Kpad, bias, bzs, C, ldc, czoff, K, (NT0),  \
                    act, aux, ldaux, nm, nr, ng, nb, gzs);                           \
            } while (0)
        if (full > 0) LAUNCH_GEMM(4, full, 0);
        if      (rem == 1) LAUNCH_GEMM(1, 1, full * 4);
        else if (rem == 2) LAUNCH_GEMM(2, 1, full * 4);
        else if (rem == 3) LAUNCH_GEMM(3, 1, full * 4);
        #undef LAUNCH_GEMM
    };

    // weight conversion (f32 -> transposed, K-padded f16)
    prep(base_W0, WtB0, 128, 400, 128, 1);
    prep(base_W1, WtB1, 400, 256, 416, 1);
    prep(em_W0,  WtE0, 128, 400, 128, 1);
    prep(gat_W0, WtG0, 400, 256, 416, 1);
    prep(gat_W1, WtG1, 256, 256, 256, 1);
    prep(sel_W,  WtS,  256,  64, 256, 3);
    prep(cond_W, WtC,   64, 256,  64, 3);
    prep(mod_W,  WtM,  256, 256, 256, 8);

    // base MLP: out = relu(x@W0+b0)@W1+b1
    gemm(x,   128, 0, WtB0, 0, 128, 128, base_b0, 0, t0,   400, 0, 400, 1, 1, nullptr, 0, nullptr, nullptr, nullptr, nullptr, 0);
    gemm(t0,  400, 0, WtB1, 0, 400, 416, base_b1, 0, outb, 256, 0, 256, 1, 0, nullptr, 0, nullptr, nullptr, nullptr, nullptr, 0);
    // embedding chain (t0 reused as scratch)
    gemm(embi,128, 0, WtE0, 0, 128, 128, em_b0,  0, t0,   400, 0, 400, 1, 1, nullptr, 0, nullptr, nullptr, nullptr, nullptr, 0);
    gemm(t0,  400, 0, WtG0, 0, 400, 416, gat_b0, 0, t2,   256, 0, 256, 1, 1, nullptr, 0, nullptr, nullptr, nullptr, nullptr, 0);
    gemm(t2,  256, 0, WtG1, 0, 256, 256, gat_b1, 0, embv, 256, 0, 256, 1, 0, nullptr, 0, nullptr, nullptr, nullptr, nullptr, 0);
    relu_copy_kernel<<<dim3((int)(((long)B*256 + 255) / 256)), blk, 0, stream>>>(embv, si, (long)B * 256);

    // select chain (sequential over 3 layers)
    for (int i = 0; i < 3; ++i) {
        gemm(si,    256, 0, WtS + (size_t)i * 64 * 256, 0, 256, 256, sel_b + i * 64, 0,
             logit, 64, 0, 64, 1, 2, nullptr, 0, nullptr, nullptr, nullptr, nullptr, 0);
        gumbel8_kernel<<<dim3((B * 8 + 255) / 256), blk, 0, stream>>>(logit, u_sel, gsel, i, B);
        gemm(logit, 64, 0, WtC + (size_t)i * 256 * 64, 0, 64, 64, cond_b + i * 256, 0,
             si, 256, 0, 256, 1, 3, embv, 256, nullptr, nullptr, nullptr, nullptr, 0);
    }
    final_select_kernel<<<dim3(B / 8), blk, 0, stream>>>(si, selF_W, selF_b, u_fin, fsel, B);

    // batchnorm stats of out, then fused (bn1-transform)@mod_W batched GEMM
    stats_partial_kernel<<<dim3(nblk), blk, 0, stream>>>(outb, 256, 256, Sp, Qp);
    stats_final_kernel<<<dim3(1), blk, 0, stream>>>(Sp, Qp, nblk, 256, mean1, rstd1, 1.0f / B);
    gemm(outb, 256, 0, WtM, (long)256 * 256, 256, 256, mod_b, 256,
         hbuf, 2048, 256, 256, 8, 0, nullptr, 0, mean1, rstd1, bn1_g, bn1_b, 256);

    // batchnorm stats of h, then per-sample cascade + head
    stats_partial_kernel<<<dim3(nblk), blk, 0, stream>>>(hbuf, 2048, 2048, Sp, Qp);
    stats_final_kernel<<<dim3(8), blk, 0, stream>>>(Sp, Qp, nblk, 2048, mean2, rstd2, 1.0f / B);
    cascade_kernel<<<dim3(B / 8), blk, 0, stream>>>(hbuf, mean2, rstd2, bn2_g, bn2_b,
                                                    gsel, fsel, last_W, last_b,
                                                    (float*)d_out, B);
}
